// JAISAttention_68135361184535
// MI455X (gfx1250) — matrix operand order repaired and checked by reading, not compiled, not run on hardware
//
#include <hip/hip_runtime.h>

typedef unsigned short u16;
typedef __attribute__((ext_vector_type(16))) __bf16 v16bf;
typedef __attribute__((ext_vector_type(8)))  float  v8f;
typedef __attribute__((ext_vector_type(8)))  unsigned int v8u;
typedef __attribute__((ext_vector_type(4)))  unsigned short us4;

// ---------- helpers ----------

__device__ __forceinline__ u16 f2bf(float f) {
    unsigned x = __builtin_bit_cast(unsigned, f);
    x += 0x7FFFu + ((x >> 16) & 1u);        // round-to-nearest-even
    return (u16)(x >> 16);
}

__device__ __forceinline__ v8f wmma_bf16(v16bf a, v16bf b, v8f c) {
    // D = A(16x32 bf16) * B(32x16 bf16) + C(16x16 f32)
    return __builtin_amdgcn_wmma_f32_16x16x32_bf16(
        false, a, false, b, (short)0, c, false, false);
}

// A fragment 16x32 bf16 from row-major [*, stride] u16 storage.
// rowptr = &base[m_row * stride]; pair K layout per ISA table:
//   p<4 : K = 2p + 8*half ; p>=4 : K = 16 + 2(p-4) + 8*half
__device__ __forceinline__ v16bf load_a_frag(const u16* rowptr, int half) {
    v8u u;
#pragma unroll
    for (int p = 0; p < 4; ++p)
        u[p] = *(const unsigned*)(rowptr + 2 * p + 8 * half);
#pragma unroll
    for (int p = 0; p < 4; ++p)
        u[4 + p] = *(const unsigned*)(rowptr + 16 + 2 * p + 8 * half);
    return __builtin_bit_cast(v16bf, u);
}

// B fragment 32x16 bf16 from K-contiguous (transposed [N][K]) u16 storage.
// colptr = &base[n_col * stride]; pair K layout per ISA table (same as A):
//   v<4 : K = 2v + 8*half ; v>=4 : K = 16 + 2(v-4) + 8*half
__device__ __forceinline__ v16bf load_b_frag(const u16* colptr, int half) {
    v8u u;
#pragma unroll
    for (int v = 0; v < 8; ++v)
        u[v] = *(const unsigned*)(colptr + 2 * v + 8 * half + (v >= 4 ? 8 : 0));
    return __builtin_bit_cast(v16bf, u);
}

// ---------- fp32 -> bf16 conversion ----------

__global__ void cvt_f32_bf16(const float* __restrict__ in, u16* __restrict__ out, long n) {
    long i = ((long)blockIdx.x * blockDim.x + threadIdx.x) * 4;
    if (i + 3 < n) {
        float4 v = *(const float4*)(in + i);
        us4 r;
        r.x = f2bf(v.x); r.y = f2bf(v.y); r.z = f2bf(v.z); r.w = f2bf(v.w);
        *(us4*)(out + i) = r;
    }
}

// ---------- tiled WMMA GEMM: C[M,N] = A[M,K] * B[K,N] + bias ----------
// BM=BN=128, BK=32, 256 threads (8 waves), wave tile 32x64 (2x4 16x16 frags)
// Register double-buffered global->LDS staging + global_prefetch 2 tiles ahead.

#define GBM 128
#define GBN 128
#define GBK 32
#define GLDA 40
#define GLDB 40

template <bool WRITE_F32>
__global__ __launch_bounds__(256) void gemm_wmma(
    const u16* __restrict__ A, const u16* __restrict__ Bm,
    const float* __restrict__ bias,
    u16* __restrict__ Cb, float* __restrict__ Cf,
    int M, int N, int K)
{
    __shared__ u16 As[GBM * GLDA];
    __shared__ u16 Bs[GBN * GLDB];

    const int t = threadIdx.x;
    const int lane = t & 31;
    const int wid  = t >> 5;
    const int wm   = wid >> 1;      // 0..3 -> 32-row slab
    const int wn   = wid & 1;       // 0..1 -> 64-col slab
    const int half = lane >> 4;
    const int nloc = lane & 15;
    const int bm = blockIdx.y, bn = blockIdx.x;

    // per-thread staging coordinates (fixed across K steps)
    const int a0_row = t >> 2,            a0_c8 = (t & 3) * 8;          // i=0
    const int a1_row = (t + 256) >> 2,    a1_c8 = ((t + 256) & 3) * 8;  // i=1
    const int b0_kr  = t >> 4,            b0_c8 = (t & 15) * 8;
    const int b1_kr  = (t + 256) >> 4,    b1_c8 = ((t + 256) & 15) * 8;

    const u16* Abase = A + (long)bm * GBM * K;
    const u16* Bbase = Bm + bn * GBN;

    v8f acc[2][4] = {};
    uint4 ra0, ra1, rb0, rb1;

    const int nk = K / GBK;
    // prologue: load tile 0 into registers
    ra0 = *(const uint4*)(Abase + (long)a0_row * K + a0_c8);
    ra1 = *(const uint4*)(Abase + (long)a1_row * K + a1_c8);
    rb0 = *(const uint4*)(Bbase + (long)b0_kr * N + b0_c8);
    rb1 = *(const uint4*)(Bbase + (long)b1_kr * N + b1_c8);

    for (int kt = 0; kt < nk; ++kt) {
        // stage registers -> LDS
        *(uint4*)(&As[a0_row * GLDA + a0_c8]) = ra0;
        *(uint4*)(&As[a1_row * GLDA + a1_c8]) = ra1;
        {
            union { uint4 v4; u16 s[8]; } tb0, tb1;
            tb0.v4 = rb0; tb1.v4 = rb1;
#pragma unroll
            for (int j = 0; j < 8; ++j) Bs[(b0_c8 + j) * GLDB + b0_kr] = tb0.s[j];
#pragma unroll
            for (int j = 0; j < 8; ++j) Bs[(b1_c8 + j) * GLDB + b1_kr] = tb1.s[j];
        }
        __syncthreads();

        // issue next tile's global loads (overlap with compute)
        if (kt + 1 < nk) {
            int k0n = (kt + 1) * GBK;
            ra0 = *(const uint4*)(Abase + (long)a0_row * K + k0n + a0_c8);
            ra1 = *(const uint4*)(Abase + (long)a1_row * K + k0n + a1_c8);
            rb0 = *(const uint4*)(Bbase + (long)(k0n + b0_kr) * N + b0_c8);
            rb1 = *(const uint4*)(Bbase + (long)(k0n + b1_kr) * N + b1_c8);
        }
        if (kt + 2 < nk) {
            int k0p = (kt + 2) * GBK;
            __builtin_prefetch(Abase + (long)a0_row * K + k0p + a0_c8, 0, 1);
            __builtin_prefetch(Bbase + (long)(k0p + b0_kr) * N + b0_c8, 0, 1);
        }

        // compute from LDS
        v16bf af[2], bf[4];
#pragma unroll
        for (int mt = 0; mt < 2; ++mt)
            af[mt] = load_a_frag(&As[(wm * 32 + mt * 16 + nloc) * GLDA], half);
#pragma unroll
        for (int nt = 0; nt < 4; ++nt)
            bf[nt] = load_b_frag(&Bs[(wn * 64 + nt * 16 + nloc) * GLDB], half);
#pragma unroll
        for (int mt = 0; mt < 2; ++mt)
#pragma unroll
            for (int nt = 0; nt < 4; ++nt)
                acc[mt][nt] = wmma_bf16(af[mt], bf[nt], acc[mt][nt]);
        __syncthreads();
    }

    // epilogue: C layout -> row = r + 8*half, col = nloc within frag
#pragma unroll
    for (int mt = 0; mt < 2; ++mt) {
#pragma unroll
        for (int nt = 0; nt < 4; ++nt) {
            const int col  = bn * GBN + wn * 64 + nt * 16 + nloc;
            const int row0 = bm * GBM + wm * 32 + mt * 16 + 8 * half;
            const float bv = bias[col];
            if (WRITE_F32) {
                float* p = Cf + (long)row0 * N + col;
#pragma unroll
                for (int r = 0; r < 8; ++r)
                    p[(long)r * N] = acc[mt][nt][r] + bv;
            } else {
                u16* p = Cb + (long)row0 * N + col;
#pragma unroll
                for (int r = 0; r < 8; ++r)
                    p[(long)r * N] = f2bf(acc[mt][nt][r] + bv);
            }
        }
    }
}

// ---------- flash attention (causal + ALiBi), bf16 WMMA, f32 softmax ----------
// Block: 128 q rows (8 waves x 16 rows), iterate keys in blocks of 32.
// hd = 128. LDS: K block [32][hd] row-major, V block transposed [hd][32].
// K/V staging register double-buffered across key blocks.

#define AQT 128
#define AKT 32
#define KLD 136   // 128 + 8 pad (u16 units), 16B-aligned rows
#define VLD 40    // 32 + 8 pad
#define PLD 40

__global__ __launch_bounds__(256) void attn_fwd(
    const u16* __restrict__ qkv,   // [B,S,3H] bf16
    u16* __restrict__ attn_out,    // [B,S,H]  bf16
    int Bsz, int S, int H, int nh)
{
    __shared__ u16 Ks[AKT * KLD];
    __shared__ u16 Vs[128 * VLD];
    __shared__ u16 Ps[8 * 16 * PLD];

    const int t = threadIdx.x, lane = t & 31, wid = t >> 5;
    const int half = lane >> 4, nloc = lane & 15;
    const int qb = blockIdx.x, h = blockIdx.y, b = blockIdx.z;
    const int hd = H / nh;                 // 128
    const long H3 = 3L * H;
    const int q0 = qb * AQT + wid * 16;    // this wave's q rows

    const u16* qptr = qkv + (long)b * S * H3 + (long)h * hd;
    const u16* kptr = qptr + H;
    const u16* vptr = qptr + 2 * H;

    const float scale = 1.0f / (float)hd;              // mup: d^-1
    const float slope = exp2f(-0.5f * (float)(h + 1)); // nh=16 pow-2 slopes

    // per-thread staging coordinates
    const int s0_kr = t >> 4,         s0_c8 = (t & 15) * 8;
    const int s1_kr = (t + 256) >> 4, s1_c8 = ((t + 256) & 15) * 8;

    // preload Q fragments: 4 frags along hd (16x32 each)
    v16bf qf[4];
    {
        const u16* qrow = qptr + (long)(q0 + nloc) * H3;
#pragma unroll
        for (int d = 0; d < 4; ++d)
            qf[d] = load_a_frag(qrow + d * 32, half);
    }

    v8f o[8] = {};
    float m_run[8], l_run[8];
#pragma unroll
    for (int r = 0; r < 8; ++r) { m_run[r] = -3.0e38f; l_run[r] = 0.0f; }

    const int nkb    = (qb * AQT + AQT + AKT - 1) / AKT;  // whole block
    const int my_nkb = (q0 + 16 + AKT - 1) / AKT;         // this wave

    // prologue: load key-block 0 into registers
    uint4 rk0, rk1, rv0, rv1;
    rk0 = *(const uint4*)(kptr + (long)s0_kr * H3 + s0_c8);
    rk1 = *(const uint4*)(kptr + (long)s1_kr * H3 + s1_c8);
    rv0 = *(const uint4*)(vptr + (long)s0_kr * H3 + s0_c8);
    rv1 = *(const uint4*)(vptr + (long)s1_kr * H3 + s1_c8);

    for (int kb = 0; kb < nkb; ++kb) {
        const int kbase = kb * AKT;
        __syncthreads();   // previous iteration's readers done
        // stage registers -> LDS (K row-major, V transposed)
        *(uint4*)(&Ks[s0_kr * KLD + s0_c8]) = rk0;
        *(uint4*)(&Ks[s1_kr * KLD + s1_c8]) = rk1;
        {
            union { uint4 v4; u16 s[8]; } tv0, tv1;
            tv0.v4 = rv0; tv1.v4 = rv1;
#pragma unroll
            for (int j = 0; j < 8; ++j) Vs[(s0_c8 + j) * VLD + s0_kr] = tv0.s[j];
#pragma unroll
            for (int j = 0; j < 8; ++j) Vs[(s1_c8 + j) * VLD + s1_kr] = tv1.s[j];
        }
        __syncthreads();

        // issue next key block's global loads (overlap with compute)
        if (kb + 1 < nkb) {
            const long nb = (long)(kbase + AKT);
            rk0 = *(const uint4*)(kptr + (nb + s0_kr) * H3 + s0_c8);
            rk1 = *(const uint4*)(kptr + (nb + s1_kr) * H3 + s1_c8);
            rv0 = *(const uint4*)(vptr + (nb + s0_kr) * H3 + s0_c8);
            rv1 = *(const uint4*)(vptr + (nb + s1_kr) * H3 + s1_c8);
        }

        if (kb >= my_nkb) continue;    // wave-uniform; barriers stay convergent

        // ---- scores: two 16x16 tiles (32 keys), K-reduce over hd in 4 WMMAs
        v8f s[2];
#pragma unroll
        for (int c = 0; c < 2; ++c) {
            v8f a = {};
#pragma unroll
            for (int d = 0; d < 4; ++d) {
                // B = K^T slice: col = key (c*16+nloc), K-dim = hd pairs (contig)
                v16bf bk = load_b_frag(&Ks[(c * 16 + nloc) * KLD + d * 32], half);
                a = wmma_bf16(qf[d], bk, a);
            }
            s[c] = a;
        }

        // ---- ALiBi + causal mask + online softmax (row = r + 8*half)
#pragma unroll
        for (int r = 0; r < 8; ++r) {
            int i = q0 + r + 8 * half;
#pragma unroll
            for (int c = 0; c < 2; ++c) {
                int j = kbase + c * 16 + nloc;
                float v = s[c][r] * scale + slope * (float)(j - i);
                if (j > i) v = -3.0e38f;
                s[c][r] = v;
            }
            float loc = fmaxf(s[0][r], s[1][r]);
            loc = fmaxf(loc, __shfl_xor(loc, 1, 32));
            loc = fmaxf(loc, __shfl_xor(loc, 2, 32));
            loc = fmaxf(loc, __shfl_xor(loc, 4, 32));
            loc = fmaxf(loc, __shfl_xor(loc, 8, 32));
            float mn = fmaxf(m_run[r], loc);
            float alpha = __expf(m_run[r] - mn);
            float p0 = __expf(s[0][r] - mn);
            float p1 = __expf(s[1][r] - mn);
            s[0][r] = p0; s[1][r] = p1;
            float rs = p0 + p1;
            rs += __shfl_xor(rs, 1, 32);
            rs += __shfl_xor(rs, 2, 32);
            rs += __shfl_xor(rs, 4, 32);
            rs += __shfl_xor(rs, 8, 32);
            l_run[r] = l_run[r] * alpha + rs;
            m_run[r] = mn;
#pragma unroll
            for (int f = 0; f < 8; ++f) o[f][r] *= alpha;
        }

        // ---- P: C-layout -> A-layout via per-wave LDS round trip
        u16* P = &Ps[wid * 16 * PLD];
#pragma unroll
        for (int r = 0; r < 8; ++r) {
            int row = r + 8 * half;
            P[row * PLD + nloc]      = f2bf(s[0][r]);
            P[row * PLD + 16 + nloc] = f2bf(s[1][r]);
        }
        asm volatile("s_wait_dscnt 0x0" ::: "memory");  // intra-wave LDS RAW

        v16bf pf = load_a_frag(&P[nloc * PLD], half);
#pragma unroll
        for (int f = 0; f < 8; ++f) {
            v16bf vf = load_b_frag(&Vs[(f * 16 + nloc) * VLD], half);
            o[f] = wmma_bf16(pf, vf, o[f]);
        }
    }

    // ---- normalize + store bf16 [B,S,H]
#pragma unroll
    for (int r = 0; r < 8; ++r) {
        const float inv = 1.0f / l_run[r];
        const int row = q0 + r + 8 * half;
        u16* p = attn_out + ((long)b * S + row) * H + h * hd + nloc;
#pragma unroll
        for (int f = 0; f < 8; ++f)
            p[f * 16] = f2bf(o[f][r] * inv);
    }
}

// ---------- launch ----------

extern "C" void kernel_launch(void* const* d_in, const int* in_sizes, int n_in,
                              void* d_out, int out_size, void* d_ws, size_t ws_size,
                              hipStream_t stream) {
    const float* hs     = (const float*)d_in[0];
    const float* w_qkv  = (const float*)d_in[1];
    const float* b_qkv  = (const float*)d_in[2];
    const float* w_proj = (const float*)d_in[3];
    const float* b_proj = (const float*)d_in[4];
    float* out = (float*)d_out;

    const int B = 2, S = 2048, H = 2048, NH = 16;
    const int M  = B * S;        // 4096
    const int N1 = 3 * H;        // 6144
    const long nHS = (long)B * S * H;      // 8,388,608
    const long nWQ = (long)H * 3 * H;      // 12,582,912
    const long nQK = (long)B * S * 3 * H;  // 25,165,824
    const long nWP = (long)H * H;          // 4,194,304

    char* ws = (char*)d_ws;
    u16* hsb    = (u16*)ws; ws += nHS * 2;
    u16* wqkvb  = (u16*)ws; ws += nWQ * 2;
    u16* qkvb   = (u16*)ws; ws += nQK * 2;
    u16* attnb  = (u16*)ws; ws += nHS * 2;
    u16* wprojb = (u16*)ws; ws += nWP * 2;

    // fp32 -> bf16 conversions (all sizes divisible by 1024)
    cvt_f32_bf16<<<(unsigned)(nHS / 4 / 256), 256, 0, stream>>>(hs,     hsb,    nHS);
    cvt_f32_bf16<<<(unsigned)(nWQ / 4 / 256), 256, 0, stream>>>(w_qkv,  wqkvb,  nWQ);
    cvt_f32_bf16<<<(unsigned)(nWP / 4 / 256), 256, 0, stream>>>(w_proj, wprojb, nWP);

    // qkv = hs @ w_qkv + b_qkv  -> bf16
    gemm_wmma<false><<<dim3(N1 / GBN, M / GBM), 256, 0, stream>>>(
        hsb, wqkvb, b_qkv, qkvb, nullptr, M, N1, H);

    // flash attention -> bf16
    attn_fwd<<<dim3(S / AQT, NH, B), 256, 0, stream>>>(qkvb, attnb, B, S, H, NH);

    // out = attn @ w_proj + b_proj -> f32
    gemm_wmma<true><<<dim3(H / GBN, M / GBM), 256, 0, stream>>>(
        attnb, wprojb, b_proj, nullptr, out, M, H, H);
}
